// AttentionBlock_61220463837659
// MI455X (gfx1250) — matrix operand order repaired and checked by reading, not compiled, not run on hardware
//
#include <hip/hip_runtime.h>

typedef __attribute__((ext_vector_type(16))) _Float16 v16h;
typedef __attribute__((ext_vector_type(8)))  float    v8f;

union VU { v16h v; unsigned int u[8]; };

#define Cc     256
#define Ntok   4096
#define Bb     4

// ---- gfx1250 async global->LDS copy (ASYNCcnt-tracked), per-lane 16B ----
__device__ __forceinline__ void async_b128(void* lds_ptr, const void* gptr) {
    unsigned loff = (unsigned)(unsigned long long)lds_ptr;   // LDS aperture: addr[31:0] = LDS byte offset
    asm volatile("global_load_async_to_lds_b128 %0, %1, off"
                 :: "v"(loff), "v"(gptr) : "memory");
}
__device__ __forceinline__ void wait_async0() {
    asm volatile("s_wait_asynccnt 0" ::: "memory");
}

// ---------------------------------------------------------------- GroupNorm stats
__global__ void gn_stats_kernel(const float* __restrict__ x, float* __restrict__ stats) {
    int bg = blockIdx.x;                             // 0..127  (b*32 + g)
    const float* base = x + (size_t)bg * 8 * Ntok;   // 8 channels * 4096 contiguous
    float s = 0.f, ss = 0.f;
    for (int i = threadIdx.x; i < 8 * Ntok; i += 256) {
        float v = base[i]; s += v; ss += v * v;
    }
    __shared__ float sh[512];
    sh[threadIdx.x] = s; sh[256 + threadIdx.x] = ss;
    __syncthreads();
    for (int off = 128; off > 0; off >>= 1) {
        if ((int)threadIdx.x < off) {
            sh[threadIdx.x]       += sh[threadIdx.x + off];
            sh[256 + threadIdx.x] += sh[256 + threadIdx.x + off];
        }
        __syncthreads();
    }
    if (threadIdx.x == 0) {
        float mean = sh[0] * (1.f / 32768.f);
        float var  = sh[256] * (1.f / 32768.f) - mean * mean;
        stats[2 * bg]     = mean;
        stats[2 * bg + 1] = rsqrtf(var + 1e-5f);
    }
}

// ---------------------------------------------------------------- GroupNorm apply + transpose to [B][N][C] f16
__global__ void gn_apply_kernel(const float* __restrict__ x,
                                const float* __restrict__ gn_w,
                                const float* __restrict__ gn_b,
                                const float* __restrict__ stats,
                                _Float16* __restrict__ h16) {
    int tid = blockIdx.x * 256 + threadIdx.x;       // over b*n*c, c fastest
    int c  = tid & 255;
    int bn = tid >> 8;
    int n  = bn & (Ntok - 1);
    int b  = bn >> 12;
    int g  = c >> 3;
    float mean = stats[2 * (b * 32 + g)];
    float rstd = stats[2 * (b * 32 + g) + 1];
    float v = x[((size_t)(b * Cc + c)) * Ntok + n];
    float r = (v - mean) * rstd * gn_w[c] + gn_b[c];
    h16[tid] = (_Float16)r;
}

// ---------------------------------------------------------------- weight f32 -> f16 (Q rows pre-scaled by 1/sqrt(C))
__global__ void wconv_kernel(const float* __restrict__ qkv_w,
                             const float* __restrict__ proj_w,
                             _Float16* __restrict__ qw16,
                             _Float16* __restrict__ pw16) {
    int tid = blockIdx.x * 256 + threadIdx.x;       // grid covers 768*256
    if (tid < 768 * 256) {
        float v = qkv_w[tid];
        if (tid < 256 * 256) v *= 0.0625f;          // scale Q weights by C^-0.5
        qw16[tid] = (_Float16)v;
    }
    if (tid < 256 * 256) pw16[tid] = (_Float16)proj_w[tid];
}

// ---------------------------------------------------------------- 16x16 tile GEMM (A row-major MxK, B column slabs: col n at B + n*ldb)
__device__ __forceinline__ v8f gemm16x16(const _Float16* __restrict__ A, int lda,
                                         const _Float16* __restrict__ B, int ldb, int K) {
    int lane = threadIdx.x & 31;
    int l16  = lane & 15;
    int b2a  = (lane & 16) ? 8  : 0;   // A-layout K base for upper half-wave
    int bb   = (lane & 16) ? 16 : 0;   // unused; B loads use the b2a k base as well
    const _Float16* Ar = A + (size_t)l16 * lda;
    const _Float16* Bc = B + (size_t)l16 * ldb;
    v8f acc = {};
    for (int kc = 0; kc < K; kc += 32) {
        VU a, bv;
#pragma unroll
        for (int i = 0; i < 8; ++i) {
            int ka = kc + b2a + ((i < 4) ? 2 * i : 2 * i + 8);
            a.u[i]  = *(const unsigned int*)(Ar + ka);
            bv.u[i] = *(const unsigned int*)(Bc + ka);
        }
        acc = __builtin_amdgcn_wmma_f32_16x16x32_f16(false, a.v, false, bv.v,
                                                     (short)0, acc, false, false);
    }
    return acc;
}

// ---------------------------------------------------------------- QKV GEMM: [768x256] x [256x4096] per batch
__global__ __launch_bounds__(256) void qkv_kernel(const _Float16* __restrict__ h16,
                                                  const _Float16* __restrict__ qw16,
                                                  const float*    __restrict__ qkv_b,
                                                  _Float16* __restrict__ q,
                                                  _Float16* __restrict__ k,
                                                  _Float16* __restrict__ v) {
    int w   = blockIdx.x * 8 + (threadIdx.x >> 5);
    int b   = w / (48 * 256);
    int rem = w % (48 * 256);
    int ot  = rem >> 8;                 // 0..47  (rows of 16 in 768)
    int nt  = rem & 255;                // 0..255 (cols of 16 in 4096)
    v8f acc = gemm16x16(qw16 + (size_t)ot * 16 * Cc, Cc,
                        h16 + ((size_t)b * Ntok + nt * 16) * Cc, Cc, Cc);
    int lane  = threadIdx.x & 31;
    int l16   = lane & 15;
    int rbase = (lane & 16) ? 8 : 0;
    int n     = nt * 16 + l16;
#pragma unroll
    for (int r = 0; r < 8; ++r) {
        int o = ot * 16 + rbase + r;
        float bias = qkv_b[o];
        if (o < 256) bias *= 0.0625f;
        float val = acc[r] + bias;
        if (o < 256)      q[((size_t)b * Ntok + n) * Cc + o]          = (_Float16)val;
        else if (o < 512) k[((size_t)b * Ntok + n) * Cc + (o - 256)]  = (_Float16)val;
        else              v[((size_t)b * Cc + (o - 512)) * Ntok + n]  = (_Float16)val;
    }
}

// ---------------------------------------------------------------- flash attention: 128 q-rows/block, async double-buffered 64-key tiles
__device__ __forceinline__ void stage_kv_async(const _Float16* __restrict__ k,
                                               const _Float16* __restrict__ v,
                                               _Float16* Ks, _Float16* Vs,
                                               int b, int kt, int tid) {
    const char* kbase = (const char*)(k + ((size_t)b * Ntok + kt) * Cc);
    char* kl = (char*)Ks;
#pragma unroll
    for (int i = 0; i < 8; ++i) {                    // K tile: 32KB flat copy
        int off = (tid + 256 * i) * 16;
        async_b128(kl + off, kbase + off);
    }
#pragma unroll
    for (int i = 0; i < 8; ++i) {                    // V tile: [256][64] channel-major
        int idx = tid + 256 * i;
        int c = idx >> 3, j = idx & 7;
        async_b128((char*)Vs + idx * 16,
                   (const char*)(v + ((size_t)b * Cc + c) * Ntok + kt + j * 8));
    }
}

__global__ __launch_bounds__(256) void attn_kernel(const _Float16* __restrict__ q,
                                                   const _Float16* __restrict__ k,
                                                   const _Float16* __restrict__ v,
                                                   _Float16* __restrict__ attnh) {
    extern __shared__ _Float16 smem[];
    _Float16* Qs  = smem;                    // [128][256]            64KB
    _Float16* KsB = Qs + 128 * Cc;           // 2 x [64][256]         2x32KB
    _Float16* VsB = KsB + 2 * 64 * Cc;       // 2 x [256][64]         2x32KB
    _Float16* Ps  = VsB + 2 * Cc * 64;       // 8 waves x [16][64]    16KB

    int b    = blockIdx.x >> 5;              // 32 row-tiles per batch
    int mt   = blockIdx.x & 31;
    int row0 = mt * 128;
    int tid  = threadIdx.x;
    int wave = tid >> 5;
    int lane = tid & 31;
    int l16  = lane & 15;
    int b2a  = (lane & 16) ? 8  : 0;
    int bb   = (lane & 16) ? 16 : 0;
    int rbase = (lane & 16) ? 8 : 0;

    // prologue: async-stage Q tile (64KB) and first K/V tile
    {
        const char* qbase = (const char*)(q + ((size_t)b * Ntok + row0) * Cc);
#pragma unroll
        for (int i = 0; i < 16; ++i) {
            int off = (tid + 256 * i) * 16;
            async_b128((char*)Qs + off, qbase + off);
        }
        stage_kv_async(k, v, KsB, VsB, b, 0, tid);
        wait_async0();
    }
    __syncthreads();

    float rmax[8], rsum[8];
#pragma unroll
    for (int r = 0; r < 8; ++r) { rmax[r] = -3.0e38f; rsum[r] = 0.f; }
    v8f acc[16];
    v8f z8 = {};
#pragma unroll
    for (int cc = 0; cc < 16; ++cc) acc[cc] = z8;

    _Float16* Pw = Ps + wave * 16 * 64;
    const _Float16* Qrow = Qs + (size_t)(wave * 16 + l16) * Cc;

    for (int kt = 0; kt < Ntok; kt += 64) {
        int cur = (kt >> 6) & 1;
        _Float16* Ks = KsB + cur * 64 * Cc;
        _Float16* Vs = VsB + cur * Cc * 64;

        // kick off async staging of the NEXT tile into the other buffer
        if (kt + 64 < Ntok)
            stage_kv_async(k, v, KsB + (cur ^ 1) * 64 * Cc, VsB + (cur ^ 1) * Cc * 64,
                           b, kt + 64, tid);

        // S = Q * K^T for four 16-key column tiles
        v8f s[4];
#pragma unroll
        for (int j = 0; j < 4; ++j) s[j] = z8;
        for (int kc = 0; kc < Cc; kc += 32) {
            VU a;
#pragma unroll
            for (int i = 0; i < 8; ++i)
                a.u[i] = *(const unsigned int*)(Qrow + kc + b2a + ((i < 4) ? 2 * i : 2 * i + 8));
#pragma unroll
            for (int j = 0; j < 4; ++j) {
                const _Float16* Kc = Ks + (size_t)(j * 16 + l16) * Cc;
                VU bv;
#pragma unroll
                for (int i = 0; i < 8; ++i)
                    bv.u[i] = *(const unsigned int*)(Kc + kc + b2a + ((i < 4) ? 2 * i : 2 * i + 8));
                s[j] = __builtin_amdgcn_wmma_f32_16x16x32_f16(false, a.v, false, bv.v,
                                                              (short)0, s[j], false, false);
            }
        }

        // online softmax (rows r+rbase live in this half-wave; columns across 16 lanes)
#pragma unroll
        for (int r = 0; r < 8; ++r) {
            float m = fmaxf(fmaxf(s[0][r], s[1][r]), fmaxf(s[2][r], s[3][r]));
            m = fmaxf(m, __shfl_xor(m, 1, 32));
            m = fmaxf(m, __shfl_xor(m, 2, 32));
            m = fmaxf(m, __shfl_xor(m, 4, 32));
            m = fmaxf(m, __shfl_xor(m, 8, 32));
            float nm = fmaxf(rmax[r], m);
            float f  = __expf(rmax[r] - nm);
            rmax[r]  = nm;
            rsum[r] *= f;
#pragma unroll
            for (int cc = 0; cc < 16; ++cc) acc[cc][r] *= f;
            float ps = 0.f;
#pragma unroll
            for (int j = 0; j < 4; ++j) {
                float p = __expf(s[j][r] - nm);
                ps += p;
                Pw[(rbase + r) * 64 + j * 16 + l16] = (_Float16)p;   // D-layout -> LDS
            }
            rsum[r] += ps;
        }
        asm volatile("s_wait_dscnt 0" ::: "memory");   // P RAW through LDS (same wave)

        // O += P * V   (K = 64 as two 32-chunks)
        {
            const _Float16* Pr = Pw + (size_t)l16 * 64;
            VU pa0, pa1;
#pragma unroll
            for (int i = 0; i < 8; ++i) {
                int kk = b2a + ((i < 4) ? 2 * i : 2 * i + 8);
                pa0.u[i] = *(const unsigned int*)(Pr + kk);
                pa1.u[i] = *(const unsigned int*)(Pr + 32 + kk);
            }
#pragma unroll
            for (int cc = 0; cc < 16; ++cc) {
                const _Float16* Vc = Vs + (size_t)(cc * 16 + l16) * 64;
                VU b0, b1;
#pragma unroll
                for (int i = 0; i < 8; ++i) {
                    b0.u[i] = *(const unsigned int*)(Vc + b2a + ((i < 4) ? 2 * i : 2 * i + 8));
                    b1.u[i] = *(const unsigned int*)(Vc + 32 + b2a + ((i < 4) ? 2 * i : 2 * i + 8));
                }
                acc[cc] = __builtin_amdgcn_wmma_f32_16x16x32_f16(false, pa0.v, false, b0.v,
                                                                 (short)0, acc[cc], false, false);
                acc[cc] = __builtin_amdgcn_wmma_f32_16x16x32_f16(false, pa1.v, false, b1.v,
                                                                 (short)0, acc[cc], false, false);
            }
        }

        wait_async0();       // own next-tile async copies have landed in LDS
        __syncthreads();     // everyone done reading cur + everyone's copies visible
    }

    // finalize: divide by row sums, store [B][N][C] f16
#pragma unroll
    for (int r = 0; r < 8; ++r) {
        float su = rsum[r];
        su += __shfl_xor(su, 1, 32);
        su += __shfl_xor(su, 2, 32);
        su += __shfl_xor(su, 4, 32);
        su += __shfl_xor(su, 8, 32);
        float inv = 1.f / su;
        int row = row0 + wave * 16 + rbase + r;
#pragma unroll
        for (int cc = 0; cc < 16; ++cc)
            attnh[((size_t)b * Ntok + row) * Cc + cc * 16 + l16] = (_Float16)(acc[cc][r] * inv);
    }
}

// ---------------------------------------------------------------- proj GEMM + bias + residual
__global__ __launch_bounds__(256) void proj_kernel(const _Float16* __restrict__ attnh,
                                                   const _Float16* __restrict__ pw16,
                                                   const float*    __restrict__ proj_b,
                                                   const float*    __restrict__ x,
                                                   float* __restrict__ out) {
    int w   = blockIdx.x * 8 + (threadIdx.x >> 5);
    int b   = w / (16 * 256);
    int rem = w % (16 * 256);
    int ot  = rem >> 8;
    int nt  = rem & 255;
    v8f acc = gemm16x16(pw16 + (size_t)ot * 16 * Cc, Cc,
                        attnh + ((size_t)b * Ntok + nt * 16) * Cc, Cc, Cc);
    int lane  = threadIdx.x & 31;
    int l16   = lane & 15;
    int rbase = (lane & 16) ? 8 : 0;
    int n     = nt * 16 + l16;
#pragma unroll
    for (int r = 0; r < 8; ++r) {
        int o = ot * 16 + rbase + r;
        size_t idx = ((size_t)b * Cc + o) * Ntok + n;
        out[idx] = x[idx] + acc[r] + proj_b[o];
    }
}

// ---------------------------------------------------------------- launcher
extern "C" void kernel_launch(void* const* d_in, const int* in_sizes, int n_in,
                              void* d_out, int out_size, void* d_ws, size_t ws_size,
                              hipStream_t stream) {
    const float* x      = (const float*)d_in[0];
    const float* gn_w   = (const float*)d_in[1];
    const float* gn_b   = (const float*)d_in[2];
    const float* qkv_w  = (const float*)d_in[3];
    const float* qkv_b  = (const float*)d_in[4];
    const float* proj_w = (const float*)d_in[5];
    const float* proj_b = (const float*)d_in[6];
    float* out = (float*)d_out;

    char* ws = (char*)d_ws;
    float*    stats = (float*)   (ws + 0);          //   1 KB
    _Float16* qw16  = (_Float16*)(ws + 1024);       // 384 KB
    _Float16* pw16  = (_Float16*)(ws + 394240);     // 128 KB
    _Float16* h16   = (_Float16*)(ws + 525312);     //   8 MB
    _Float16* qbuf  = (_Float16*)(ws + 8913920);    //   8 MB
    _Float16* kbuf  = (_Float16*)(ws + 17302528);   //   8 MB
    _Float16* vbuf  = (_Float16*)(ws + 25691136);   //   8 MB
    _Float16* atth  = (_Float16*)(ws + 34079744);   //   8 MB   (total ~40.5 MB)

    gn_stats_kernel<<<128, 256, 0, stream>>>(x, stats);
    gn_apply_kernel<<<16384, 256, 0, stream>>>(x, gn_w, gn_b, stats, h16);
    wconv_kernel<<<768, 256, 0, stream>>>(qkv_w, proj_w, qw16, pw16);
    qkv_kernel<<<6144, 256, 0, stream>>>(h16, qw16, qkv_b, qbuf, kbuf, vbuf);
    size_t lds = (size_t)(128 * 256 + 2 * 64 * 256 + 2 * 256 * 64 + 8 * 16 * 64) * sizeof(_Float16); // 208 KB
    attn_kernel<<<128, 256, lds, stream>>>(qbuf, kbuf, vbuf, atth);
    proj_kernel<<<2048, 256, 0, stream>>>(atth, pw16, proj_b, x, out);
}
